// ProbOhemCrossEntropy2d_4114578669600
// MI455X (gfx1250) — compile-verified, matrix operand order checked by reading, not run on hardware
//
#include <hip/hip_runtime.h>

#define IGNORE_LABEL 255
#define THRESH_F 0.7f
#define MIN_KEPT 262144u

#define NCLS 19
#define HWPIX (768 * 768)          /* 589824 */
#define NPIX  (8 * HWPIX)          /* 4718592 */
#define NB3   1024                 /* blocks in k_reduce */

typedef __attribute__((ext_vector_type(16))) _Float16 v16h;
typedef __attribute__((ext_vector_type(8)))  float    v8f;

struct Ctl {
    unsigned int hist[4][256];
    unsigned int num_valid;
    unsigned int prefix;
    unsigned int rank;
    float        threshold;
};

// ---------------------------------------------------------------- init ------
__global__ void k_init(Ctl* __restrict__ ctl) {
    int t = threadIdx.x;                      // 1024 threads cover 4*256 bins
    ((unsigned int*)ctl->hist)[t] = 0u;
    if (t == 0) {
        ctl->num_valid = 0u;
        ctl->prefix    = 0u;
        ctl->rank      = MIN_KEPT;            // 1-indexed k-th smallest
        ctl->threshold = 0.0f;
    }
}

// ------------------------------------------------------- main softmax ------
__global__ void __launch_bounds__(256)
k_main(const float* __restrict__ pred, const int* __restrict__ target,
       float* __restrict__ maskp, float* __restrict__ negl,
       Ctl* __restrict__ ctl) {
    const int n = blockIdx.x * 256 + threadIdx.x;     // exact cover of NPIX

    __shared__ int tgt[256];
#if defined(__gfx1250__) && __has_builtin(__builtin_amdgcn_global_load_async_to_lds_b32)
    // CDNA5 async global->LDS path (ASYNCcnt), one b32 per lane.
    __builtin_amdgcn_global_load_async_to_lds_b32(
        (__attribute__((address_space(1))) int*)(target + n),
        (__attribute__((address_space(3))) int*)(&tgt[threadIdx.x]),
        0, 0);
    asm volatile("s_wait_asynccnt 0" ::: "memory");
    __syncthreads();
    const int t = tgt[threadIdx.x];
#else
    tgt[threadIdx.x] = target[n];
    __syncthreads();
    const int t = tgt[threadIdx.x];
#endif

    const int b  = n / HWPIX;
    const int hw = n - b * HWPIX;
    const float* p = pred + (size_t)b * NCLS * HWPIX + hw;

    const bool valid = (t != IGNORE_LABEL);
    const int  tc    = valid ? t : 0;

    float x[NCLS];
    float m = -__builtin_inff();
#pragma unroll
    for (int c = 0; c < NCLS; ++c) {
        x[c] = p[(size_t)c * HWPIX];
        m = fmaxf(m, x[c]);
    }
    float xt = x[0];
#pragma unroll
    for (int c = 1; c < NCLS; ++c)
        if (c == tc) xt = x[c];

    float s = 0.0f;
#pragma unroll
    for (int c = 0; c < NCLS; ++c)
        s += __expf(x[c] - m);

    const float lse = m + __logf(s);
    const float nl  = fmaxf(lse - xt, 0.0f);          // -log p_t  (>= 0)

    maskp[n] = valid ? __expf(-nl) : 1.0f;            // gt-class probability
    negl[n]  = valid ? nl : -1.0f;                    // sign bit == invalid

    // valid count: ballot + one atomic per wave
    unsigned long long ball = __ballot(valid);
    if ((threadIdx.x & 31) == 0)
        atomicAdd(&ctl->num_valid, (unsigned int)__popcll(ball));
}

// ----------------------------------------------- radix-select histogram ----
__global__ void __launch_bounds__(256)
k_hist(const unsigned int* __restrict__ bits, Ctl* __restrict__ ctl, int pass) {
    __shared__ unsigned int h[256];
    h[threadIdx.x] = 0u;
    __syncthreads();

    const unsigned int pfx = ctl->prefix;
    const int stride = gridDim.x * 256;
    for (int i = blockIdx.x * 256 + threadIdx.x; i < NPIX; i += stride) {
        const unsigned int v = bits[i];
        const bool match = (pass == 0) || ((v >> (32 - 8 * pass)) == pfx);
        if (match)
            atomicAdd(&h[(v >> (24 - 8 * pass)) & 255u], 1u);
    }
    __syncthreads();
    atomicAdd(&ctl->hist[pass][threadIdx.x], h[threadIdx.x]);
}

__global__ void k_scan(Ctl* __restrict__ ctl, int pass) {
    if (threadIdx.x != 0) return;
    const unsigned int rank = ctl->rank;
    unsigned int cum = 0u, d = 255u;
    for (unsigned int i = 0; i < 256u; ++i) {
        const unsigned int c = ctl->hist[pass][i];
        if (cum + c >= rank) { d = i; break; }
        cum += c;
    }
    ctl->prefix = (ctl->prefix << 8) | d;
    ctl->rank   = rank - cum;
    if (pass == 3) {
        const float kth = __uint_as_float(ctl->prefix);
        ctl->threshold  = fmaxf(kth, THRESH_F);
    }
}

// ------------------------------------------- masked sum via WMMA reduce ----
__global__ void __launch_bounds__(256)
k_reduce(const float* __restrict__ maskp, const float* __restrict__ negl,
         const Ctl* __restrict__ ctl,
         float* __restrict__ bsum, float* __restrict__ bcnt) {
    const float thr     = ctl->threshold;
    const bool  do_ohem = (ctl->num_valid >= MIN_KEPT);

    float lsum = 0.0f, lcnt = 0.0f;
    const int stride = gridDim.x * 256;
    for (int i = blockIdx.x * 256 + threadIdx.x; i < NPIX; i += stride) {
        const float nl    = negl[i];
        const bool  valid = (nl >= 0.0f);
        bool keep = valid;
        if (do_ohem) keep = valid && (maskp[i] <= thr);
        if (keep) { lsum += nl; lcnt += 1.0f; }
    }

    // Per-wave reduction on the matrix engine: A holds one value per lane
    // (slot 0 of its 16 halves), B = all-ones.  D = A x B  =>  every column of
    // D is the vector of row-sums of A, so sum of the 8 f32 accumulator VGPRs
    // at lanes 0 and 16 (M=0..7 and M=8..15, N=0) is the wave-wide total.
    v16h a, ones;
#pragma unroll
    for (int j = 0; j < 16; ++j) { a[j] = (_Float16)0.0f; ones[j] = (_Float16)1.0f; }
    v8f cz;
#pragma unroll
    for (int j = 0; j < 8; ++j) cz[j] = 0.0f;

    a[0] = (_Float16)lsum;
    v8f ds = __builtin_amdgcn_wmma_f32_16x16x32_f16(false, a, false, ones,
                                                    (short)0, cz, false, false);
    a[0] = (_Float16)lcnt;
    v8f dc = __builtin_amdgcn_wmma_f32_16x16x32_f16(false, a, false, ones,
                                                    (short)0, cz, false, false);
    float qs = 0.0f, qc = 0.0f;
#pragma unroll
    for (int j = 0; j < 8; ++j) { qs += ds[j]; qc += dc[j]; }
    const float wsum = __shfl(qs, 0, 32) + __shfl(qs, 16, 32);
    const float wcnt = __shfl(qc, 0, 32) + __shfl(qc, 16, 32);

    __shared__ float wss[8], wcs[8];
    const int wave = threadIdx.x >> 5;
    if ((threadIdx.x & 31) == 0) { wss[wave] = wsum; wcs[wave] = wcnt; }
    __syncthreads();
    if (threadIdx.x == 0) {
        float s = 0.0f, c = 0.0f;
        for (int w = 0; w < 8; ++w) { s += wss[w]; c += wcs[w]; }
        bsum[blockIdx.x] = s;
        bcnt[blockIdx.x] = c;
    }
}

// ------------------------------------------------------------- finalize ----
__global__ void k_final(const float* __restrict__ bsum,
                        const float* __restrict__ bcnt,
                        float* __restrict__ out) {
    __shared__ float ss[256], sc[256];
    const int t = threadIdx.x;
    float s = 0.0f, c = 0.0f;
#pragma unroll
    for (int j = 0; j < NB3 / 256; ++j) {
        s += bsum[t + j * 256];
        c += bcnt[t + j * 256];
    }
    ss[t] = s; sc[t] = c;
    __syncthreads();
    for (int off = 128; off > 0; off >>= 1) {
        if (t < off) { ss[t] += ss[t + off]; sc[t] += sc[t + off]; }
        __syncthreads();
    }
    if (t == 0) {
        const float denom = fmaxf(sc[0], 1.0f);
        out[0] = ss[0] / denom;
    }
}

// -------------------------------------------------------------- launch -----
extern "C" void kernel_launch(void* const* d_in, const int* in_sizes, int n_in,
                              void* d_out, int out_size, void* d_ws, size_t ws_size,
                              hipStream_t stream) {
    const float* pred   = (const float*)d_in[0];
    const int*   target = (const int*)d_in[1];
    float*       out    = (float*)d_out;

    char*  ws    = (char*)d_ws;
    float* maskp = (float*)ws;                                   // NPIX f32
    float* negl  = (float*)(ws + (size_t)NPIX * 4);              // NPIX f32
    Ctl*   ctl   = (Ctl*)  (ws + (size_t)2 * NPIX * 4);          // ~4.1 KB
    float* bsum  = (float*)(ws + (size_t)2 * NPIX * 4 + 8192);   // NB3 f32
    float* bcnt  = bsum + NB3;                                   // NB3 f32

    k_init<<<1, 1024, 0, stream>>>(ctl);
    k_main<<<NPIX / 256, 256, 0, stream>>>(pred, target, maskp, negl, ctl);
    for (int pass = 0; pass < 4; ++pass) {
        k_hist<<<512, 256, 0, stream>>>((const unsigned int*)maskp, ctl, pass);
        k_scan<<<1, 32, 0, stream>>>(ctl, pass);
    }
    k_reduce<<<NB3, 256, 0, stream>>>(maskp, negl, ctl, bsum, bcnt);
    k_final<<<1, 256, 0, stream>>>(bsum, bcnt, out);
}
